// GroupAttention_37168646980243
// MI455X (gfx1250) — compile-verified, matrix operand order checked by reading, not run on hardware
//
#include <hip/hip_runtime.h>
#include <hip/hip_bf16.h>

// ---------------- problem constants ----------------
constexpr int S   = 196;     // sequence length
constexpr int Bt  = 256;     // batch
constexpr int E   = 1152;    // embed dim
constexpr int D   = 128;     // head dim
constexpr int SP  = 208;     // S padded to multiple of 16 (13 tiles)
constexpr int MT  = 13;      // m tiles
constexpr int SCP = 209;     // score row stride (odd -> conflict-free LDS)
constexpr int NT  = 512;     // threads per block (16 waves)
constexpr float SCALE = 0.08838834764831845f; // 1/sqrt(128)

typedef __attribute__((ext_vector_type(16))) _Float16 v16h;
typedef __attribute__((ext_vector_type(8)))  _Float16 v8h;
typedef __attribute__((ext_vector_type(8)))  float    v8f;

// LDS layout (bytes)
constexpr int QH_HALFS  = SP * D;                // 26624 halfs
constexpr int QK_BYTES  = QH_HALFS * 2;          // one f16 buffer = 53248 B
constexpr int SC_OFF    = 2 * QK_BYTES;          // 106496
constexpr int SC_FLOATS = SP * SCP;              // 43472 (full padded rows)
constexpr int SMEM_BYTES = SC_OFF + (SC_FLOATS + 2 * S) * 4; // 281,952 B

__device__ __forceinline__ v16h pack16(float4 a, float4 b, float4 c, float4 d) {
  v16h v;
  v[0]=(_Float16)a.x;  v[1]=(_Float16)a.y;  v[2]=(_Float16)a.z;  v[3]=(_Float16)a.w;
  v[4]=(_Float16)b.x;  v[5]=(_Float16)b.y;  v[6]=(_Float16)b.z;  v[7]=(_Float16)b.w;
  v[8]=(_Float16)c.x;  v[9]=(_Float16)c.y;  v[10]=(_Float16)c.z; v[11]=(_Float16)c.w;
  v[12]=(_Float16)d.x; v[13]=(_Float16)d.y; v[14]=(_Float16)d.z; v[15]=(_Float16)d.w;
  return v;
}

__global__ __launch_bounds__(NT, 1)
void group_attention_fused(const float* __restrict__ ctx,   // [S,B,E]
                           const float* __restrict__ prior, // [B,S,S]
                           const float* __restrict__ Wq,    // [D,E]
                           const float* __restrict__ bq,    // [D]
                           const float* __restrict__ Wk,    // [D,E]
                           const float* __restrict__ bk,    // [D]
                           float* __restrict__ out1,        // g_attn   [B,S,S]
                           float* __restrict__ out2)        // neibor   [B,S,S]
{
  extern __shared__ char smem[];
  _Float16* qh    = (_Float16*)smem;                 // [SP][D] f16
  _Float16* kh    = qh + QH_HALFS;                   // [SP][D] f16
  float*    sc    = (float*)(smem + SC_OFF);         // [SP][SCP] f32
  float*    Ls    = sc + SC_FLOATS;                  // [S]
  float*    diagv = Ls + S;                          // [S]

  const int b     = blockIdx.x;
  const int tid   = threadIdx.x;
  const int wave  = tid >> 5;
  const int lane  = tid & 31;
  const int lh    = lane & 15;        // 0..15
  const int hi    = (lane >> 4) & 1;  // 0/1

  // =========== Phase A0: Q,K projection via WMMA f16 (f32 accum) ===========
  // 16 waves: waves 0..7 -> Q n-tiles, waves 8..15 -> K n-tiles.
  // Padded A-matrix rows (196..207) are CLAMPED to row 195; their results are
  // written unconditionally into the padded LDS rows (in-bounds, and only ever
  // feed score rows/columns that are themselves discarded). No divergence, no
  // guards, no zero-memset needed: every [SP][D] entry is written below.
  {
    const bool isK  = (wave >> 3) != 0;
    const float* W  = isK ? Wk : Wq;
    const float* bi = isK ? bk : bq;
    _Float16* dst   = isK ? kh : qh;
    const int n0    = (wave & 7) * 16;
    const float bias = bi[n0 + lh];
    const float* wrow = W + (size_t)(n0 + lh) * E;

    for (int mt = 0; mt < MT; ++mt) {
      const int m0 = mt * 16;
      const int arowClamp = min(m0 + lh, S - 1);      // clamp padded rows
      const float* actx = ctx + ((size_t)arowClamp * Bt + b) * E;
      v8f acc = {0.f,0.f,0.f,0.f,0.f,0.f,0.f,0.f};

      for (int k0 = 0; k0 < E; k0 += 32) {
        const int kb = k0 + hi * 8;
        __builtin_prefetch(actx + kb + 64, 0, 1);     // global_prefetch_b8
        float4 f0 = *(const float4*)(actx + kb);
        float4 f1 = *(const float4*)(actx + kb + 4);
        float4 f2 = *(const float4*)(actx + kb + 16);
        float4 f3 = *(const float4*)(actx + kb + 20);
        v16h af = pack16(f0, f1, f2, f3);

        float4 g0 = *(const float4*)(wrow + kb);
        float4 g1 = *(const float4*)(wrow + kb + 4);
        float4 g2 = *(const float4*)(wrow + kb + 16);
        float4 g3 = *(const float4*)(wrow + kb + 20);
        v16h bf = pack16(g0, g1, g2, g3);

        acc = __builtin_amdgcn_wmma_f32_16x16x32_f16(
                  false, af, false, bf, (short)0, acc, false, false);
      }
      // unconditional store-back (m0 + r + hi*8 <= 207 < SP)
      const int mbase = m0 + hi * 8;
      #pragma unroll
      for (int r = 0; r < 8; ++r) {
        dst[(mbase + r) * D + n0 + lh] = (_Float16)(acc[r] + bias);
      }
    }
  }
  __syncthreads();

  // =========== Phase A1: scores = Q K^T * (1/sqrt(128)) via WMMA ===========
  for (int tile = wave; tile < MT * MT; tile += 16) {
    const int mt = tile / MT, nt = tile % MT;
    const _Float16* ap = qh + (mt * 16 + lh) * D;
    const _Float16* bp = kh + (nt * 16 + lh) * D;
    v8f acc = {0.f,0.f,0.f,0.f,0.f,0.f,0.f,0.f};
    #pragma unroll
    for (int k0 = 0; k0 < D; k0 += 32) {
      const int kb = k0 + hi * 8;
      v8h alo = *(const v8h*)(ap + kb);
      v8h ahi = *(const v8h*)(ap + kb + 16);
      v16h af = __builtin_shufflevector(alo, ahi, 0,1,2,3,4,5,6,7,8,9,10,11,12,13,14,15);
      v8h blo = *(const v8h*)(bp + kb);
      v8h bhi = *(const v8h*)(bp + kb + 16);
      v16h bf = __builtin_shufflevector(blo, bhi, 0,1,2,3,4,5,6,7,8,9,10,11,12,13,14,15);
      acc = __builtin_amdgcn_wmma_f32_16x16x32_f16(
                false, af, false, bf, (short)0, acc, false, false);
    }
    // unconditional store-back: sc is [SP][SCP], m <= 207, n <= 207 < SCP
    const int mbase = mt * 16 + hi * 8;
    const int n = nt * 16 + lh;
    #pragma unroll
    for (int r = 0; r < 8; ++r) {
      sc[(mbase + r) * SCP + n] = acc[r] * SCALE;
    }
  }
  __syncthreads();

  // =========== Phase B: row softmax in LDS (valid region only) ===========
  if (tid < S) {
    float* row = sc + tid * SCP;
    float mx = row[0];
    for (int j = 1; j < S; ++j) mx = fmaxf(mx, row[j]);
    float sum = 0.f;
    for (int j = 0; j < S; ++j) { float e = expf(row[j] - mx); row[j] = e; sum += e; }
    const float inv = 1.f / sum;
    for (int j = 0; j < S; ++j) row[j] *= inv;
  }
  __syncthreads();

  // =========== Phase C1: na = sqrt(na * na^T + 1e-9) (pairs i<=j) ===========
  for (int idx = tid; idx < S * S; idx += NT) {
    const int i = idx / S, j = idx % S;
    if (i > j) continue;
    const float x = sc[i * SCP + j];
    const float y = sc[j * SCP + i];
    const float s = sqrtf(x * y + 1e-9f);
    sc[i * SCP + j] = s;
    sc[j * SCP + i] = s;
  }
  __syncthreads();

  // =========== Phase C2: mix with prior; write output 2 (coalesced) ========
  {
    const float* pr = prior + (size_t)b * S * S;
    float* o2 = out2 + (size_t)b * S * S;
    for (int idx = tid; idx < S * S; idx += NT) {
      const int i = idx / S, j = idx % S;
      const float p = pr[idx];
      const float m = p + (1.f - p) * sc[i * SCP + j];
      sc[i * SCP + j] = m;
      o2[idx] = m;
    }
  }
  __syncthreads();

  // =========== extract diag + superdiagonal factors ===========
  if (tid < S)     diagv[tid] = sc[tid * SCP + tid];
  if (tid < S - 1) Ls[tid]    = sc[tid * SCP + tid + 1] + 1e-9f;
  __syncthreads();

  // =========== Phase D1: running products along each row (strict upper) ====
  if (tid < S) {
    float p = 1.f;
    for (int k = tid + 1; k < S; ++k) {
      p *= Ls[k - 1];
      sc[tid * SCP + k] = p;      // g[tid][k] = prod_{j=tid}^{k-1} L[j]
    }
  }
  __syncthreads();

  // =========== Phase D2: assemble g_attn; write output 1 (coalesced) =======
  {
    float* o1 = out1 + (size_t)b * S * S;
    for (int idx = tid; idx < S * S; idx += NT) {
      const int i = idx / S, j = idx % S;
      float v;
      if (i == j)      v = diagv[i];
      else if (i < j)  v = sc[i * SCP + j] + 1e-9f;
      else             v = sc[j * SCP + i] + 1e-9f;
      o1[idx] = v;
    }
  }
}

extern "C" void kernel_launch(void* const* d_in, const int* in_sizes, int n_in,
                              void* d_out, int out_size, void* d_ws, size_t ws_size,
                              hipStream_t stream) {
  const float* ctx   = (const float*)d_in[0]; // context [S,B,E]
  const float* prior = (const float*)d_in[1]; // [B,S,S]
  const float* Wq    = (const float*)d_in[2];
  const float* bq    = (const float*)d_in[3];
  const float* Wk    = (const float*)d_in[4];
  const float* bk    = (const float*)d_in[5];
  float* out1 = (float*)d_out;                        // g_attn
  float* out2 = out1 + (size_t)Bt * S * S;            // neibor_attn

  (void)d_ws; (void)ws_size; (void)in_sizes; (void)n_in; (void)out_size;

  // allow >64KB dynamic LDS (CDNA5 WGP has 320KB)
  hipFuncSetAttribute((const void*)group_attention_fused,
                      hipFuncAttributeMaxDynamicSharedMemorySize, SMEM_BYTES);

  group_attention_fused<<<dim3(Bt), dim3(NT), SMEM_BYTES, stream>>>(
      ctx, prior, Wq, bq, Wk, bk, out1, out2);
}
